// LSTMBlock_25486335934893
// MI455X (gfx1250) — matrix operand order repaired and checked by reading, not compiled, not run on hardware
//
#include <hip/hip_runtime.h>
#include <hip/hip_bf16.h>

#define TSEQ 2048
#define NB   256
#define HID  128
#define G4   512   // 4*H
#define TC   128   // timestep chunk

typedef __attribute__((ext_vector_type(16))) __bf16         v16bf;
typedef __attribute__((ext_vector_type(16))) unsigned short v16us;
typedef __attribute__((ext_vector_type(8)))  float          v8f;

// ---------------- helpers ----------------
__device__ __forceinline__ unsigned short f2bf(float f) {
  unsigned u = __float_as_uint(f);
  u += 0x7fffu + ((u >> 16) & 1u);          // round to nearest even
  return (unsigned short)(u >> 16);
}
__device__ __forceinline__ float bf2f(unsigned short b) {
  return __uint_as_float((unsigned)b << 16);
}
__device__ __forceinline__ unsigned pack2bf(float a, float b) {
  return (unsigned)f2bf(a) | ((unsigned)f2bf(b) << 16);
}
__device__ __forceinline__ float sigmoid_fast(float x) {
  return 1.0f / (1.0f + __expf(-x));
}
__device__ __forceinline__ float tanh_fast(float x) {
  return 1.0f - 2.0f / (__expf(2.0f * x) + 1.0f);
}

// A fragment: 16x32 bf16, row-major source, row stride ld (elements).
// lane row m = lane%16 ; half = lane/16.
__device__ __forceinline__ v16bf loadA(const unsigned* p32, int ldHalf, int m,
                                       int k0, int half) {
  v16us t;
#pragma unroll
  for (int v = 0; v < 8; ++v) {
    int kk = 2 * v + 16 * half;
    unsigned u = p32[m * ldHalf + ((k0 + kk) >> 1)];
    t[2 * v]     = (unsigned short)(u & 0xffffu);
    t[2 * v + 1] = (unsigned short)(u >> 16);
  }
  return __builtin_bit_cast(v16bf, t);
}

// B fragment: 32x16, B(k,n) = W[g][k], W row-major [4H][K].
__device__ __forceinline__ v16bf loadB(const unsigned* w32, int kHalf, int g,
                                       int k0, int half) {
  v16us t;
  int base = g * kHalf + (k0 >> 1) + 8 * half;
#pragma unroll
  for (int v = 0; v < 8; ++v) {
    unsigned u = w32[base + v];
    t[2 * v]     = (unsigned short)(u & 0xffffu);
    t[2 * v + 1] = (unsigned short)(u >> 16);
  }
  return __builtin_bit_cast(v16bf, t);
}

// ---------------- fp32 -> bf16 weight conversion ----------------
__global__ __launch_bounds__(256) void cvt_f32_bf16(const float* __restrict__ s,
                                                    unsigned short* __restrict__ d,
                                                    int n) {
  int i = blockIdx.x * 256 + threadIdx.x;
  if (i < n) d[i] = f2bf(s[i]);
}

// ---------------- input projection: pre = bf16(x @ W_ih^T + b) ----------------
// grid.x = TC*16 (s_local*16 + batch_tile), grid.y = dir ; 256 threads (8 waves)
// pre layout: [2][TC][B][512] bf16
template <int K, bool L1>
__global__ __launch_bounds__(256) void proj_kernel(
    const void* __restrict__ xin, const unsigned* __restrict__ wihF,
    const unsigned* __restrict__ wihB, const float* __restrict__ biasF,
    const float* __restrict__ biasB, unsigned short* __restrict__ pre,
    int chunk_start) {
  __shared__ __align__(16) unsigned ldsA[(16 * K) / 2];   // 16 x K bf16 tile

  const int tid  = threadIdx.x;
  const int dir  = blockIdx.y;
  const int sloc = blockIdx.x >> 4;
  const int bt   = blockIdx.x & 15;
  const int b0   = bt * 16;
  const int S    = chunk_start + sloc;
  const int t    = dir ? (TSEQ - 1 - S) : S;

  if (!L1) {
    const float* xf = (const float*)xin;              // [B,T,64] fp32
    int r  = tid >> 4;
    int c4 = (tid & 15) * 4;
    const float4 v = *reinterpret_cast<const float4*>(
        xf + ((size_t)(b0 + r) * TSEQ + t) * 64 + c4);
    ldsA[r * (K / 2) + (c4 >> 1) + 0] = pack2bf(v.x, v.y);
    ldsA[r * (K / 2) + (c4 >> 1) + 1] = pack2bf(v.z, v.w);
  } else {
    const unsigned short* xb = (const unsigned short*)xin;  // [B,T,256] bf16
    int r  = tid >> 4;
    int c0 = (tid & 15) * 16;
    const uint4* src = reinterpret_cast<const uint4*>(
        xb + ((size_t)(b0 + r) * TSEQ + t) * 256 + c0);
    uint4* dst = reinterpret_cast<uint4*>(&ldsA[r * (K / 2) + (c0 >> 1)]);
    dst[0] = src[0];
    dst[1] = src[1];
  }
  __syncthreads();

  const int wave = tid >> 5, lane = tid & 31;
  const int col = lane & 15, half = lane >> 4;
  const unsigned* wih = dir ? wihB : wihF;
  const float* bias   = dir ? biasB : biasF;
  const size_t prebase = ((size_t)dir * TC + sloc) * (size_t)NB * G4;

#pragma unroll
  for (int nt = 0; nt < 4; ++nt) {
    const int g0 = wave * 64 + nt * 16;
    v8f acc = {};
#pragma unroll
    for (int k0 = 0; k0 < K; k0 += 32) {
      v16bf a = loadA(ldsA, K / 2, col, k0, half);
      v16bf b = loadB(wih, K / 2, g0 + col, k0, half);
      acc = __builtin_amdgcn_wmma_f32_16x16x32_bf16(false, a, false, b,
                                                    (short)0, acc, false, false);
    }
    const float bv = bias[g0 + col];
#pragma unroll
    for (int v = 0; v < 8; ++v) {
      int m = v + 8 * half;
      pre[prebase + (size_t)(b0 + m) * G4 + g0 + col] = f2bf(acc[v] + bv);
    }
  }
}

// ---------------- recurrence (persistent per chunk) ----------------
// grid = (16 batch tiles, 2 dirs), 256 threads = 8 waves.
// Wave w owns h-columns [16w,16w+16) and gate tiles {16w, 16w+128, +256, +384}:
// i/f/g/o for its h-columns land in its own WMMA accumulators (no LDS exchange).
// w_hh B-fragments are register-resident for the whole chunk.
// pre is prefetched 1 step ahead with global_load_async_to_lds_b128 into a
// wave-private double-buffered LDS slice (ASYNCcnt is wave-local, no barriers).
template <bool L0>
__global__ __launch_bounds__(256) void recur_kernel(
    const unsigned short* __restrict__ pre, const unsigned* __restrict__ whhF,
    const unsigned* __restrict__ whhB, float* __restrict__ stateH,
    float* __restrict__ stateC, unsigned short* __restrict__ x1,
    float* __restrict__ out, int chunk_start, int last_chunk) {
  __shared__ __align__(16) unsigned short hA[2][16 * HID];  // 2 x 4KB bf16
  __shared__ __align__(16) unsigned char preL[8 * 2 * 4 * 512];  // 32KB

  const int tid  = threadIdx.x;
  const int dir  = blockIdx.y;
  const int b0   = blockIdx.x * 16;
  const int wave = tid >> 5, lane = tid & 31;
  const int col = lane & 15, half = lane >> 4;
  const int hcol = wave * 16 + col;              // owned h column (0..127)
  const unsigned* whh = dir ? whhB : whhF;

  // ---- hoist w_hh fragments into registers for the whole chunk ----
  v16bf wreg[4][4];
#pragma unroll
  for (int g = 0; g < 4; ++g)
#pragma unroll
    for (int k = 0; k < 4; ++k)
      wreg[g][k] = loadB(whh, HID / 2, g * HID + wave * 16 + col, 32 * k, half);

  // ---- load / init carried state: lane owns (rows m=v+8*half, col hcol) ----
  float hreg[8], creg[8];
  if (chunk_start == 0) {
#pragma unroll
    for (int v = 0; v < 8; ++v) { hreg[v] = 0.f; creg[v] = 0.f; }
  } else {
#pragma unroll
    for (int v = 0; v < 8; ++v) {
      int m = v + 8 * half;
      hreg[v] = stateH[((size_t)dir * NB + b0 + m) * HID + hcol];
      creg[v] = stateC[((size_t)dir * NB + b0 + m) * HID + hcol];
    }
  }
#pragma unroll
  for (int v = 0; v < 8; ++v)
    hA[0][(v + 8 * half) * HID + hcol] = f2bf(hreg[v]);

  // ---- async prefetch helpers ----
  const unsigned preLds0 = (unsigned)(uintptr_t)(void*)preL;  // LDS byte offset
  const unsigned long long preSB = (unsigned long long)(uintptr_t)pre;
  const int prow = lane >> 1, pchk = lane & 1;  // each lane moves 16B/row-chunk
  auto prefetch = [&](int s_pf, int buf) {
    unsigned ldsDst = preLds0 + wave * 4096 + buf * 2048 + prow * 32 + pchk * 16;
    size_t eb = (((size_t)dir * TC + s_pf) * NB + b0 + prow) * (size_t)G4 +
                wave * 16 + pchk * 8;
#pragma unroll
    for (int g = 0; g < 4; ++g) {
      unsigned goff = (unsigned)((eb + (size_t)g * HID) * 2);
      unsigned ldst = ldsDst + g * 512;
      asm volatile("global_load_async_to_lds_b128 %0, %1, %2"
                   :: "v"(ldst), "v"(goff), "s"(preSB) : "memory");
    }
  };
  prefetch(0, 0);   // pre for step 0 -> buffer 0
  __syncthreads();

  for (int s = 0; s < TC; ++s) {
    const int p = s & 1;
    const int sn = (s + 1 < TC) ? s + 1 : TC - 1;
    prefetch(sn, 1 - p);   // overlap next step's pre with this step's GEMM

    // ---- h @ W_hh^T : 4 gate tiles, K=128 ----
    v16bf afrag[4];
#pragma unroll
    for (int k = 0; k < 4; ++k)
      afrag[k] = loadA((const unsigned*)hA[p], HID / 2, col, 32 * k, half);
    v8f acc[4];
#pragma unroll
    for (int g = 0; g < 4; ++g) {
      v8f a = {};
#pragma unroll
      for (int k = 0; k < 4; ++k)
        a = __builtin_amdgcn_wmma_f32_16x16x32_bf16(false, afrag[k], false,
                                                    wreg[g][k], (short)0, a,
                                                    false, false);
      acc[g] = a;
    }

    // ---- wait for this step's pre slice (wave-local ASYNCcnt) ----
    asm volatile("s_wait_asynccnt 0x4" ::: "memory");
    const unsigned short* pp = (const unsigned short*)(
        preL + wave * 4096 + p * 2048);

    // ---- gates entirely in registers ----
    const int S = chunk_start + s;
    const int t = dir ? (TSEQ - 1 - S) : S;
#pragma unroll
    for (int v = 0; v < 8; ++v) {
      int m = v + 8 * half;
      float iv = acc[0][v] + bf2f(pp[0 * 256 + m * 16 + col]);
      float fv = acc[1][v] + bf2f(pp[1 * 256 + m * 16 + col]);
      float gv = acc[2][v] + bf2f(pp[2 * 256 + m * 16 + col]);
      float ov = acc[3][v] + bf2f(pp[3 * 256 + m * 16 + col]);
      iv = sigmoid_fast(iv);
      fv = sigmoid_fast(fv);
      gv = tanh_fast(gv);
      ov = sigmoid_fast(ov);
      creg[v] = fv * creg[v] + iv * gv;
      hreg[v] = ov * tanh_fast(creg[v]);
      unsigned short hb = f2bf(hreg[v]);
      hA[1 - p][m * HID + hcol] = hb;            // A matrix for step s+1
      if (L0)
        x1[((size_t)(b0 + m) * TSEQ + t) * (2 * HID) + dir * HID + hcol] = hb;
    }
    __syncthreads();   // hA[1-p] complete before anyone reads it
  }

  // persist carry across chunks
#pragma unroll
  for (int v = 0; v < 8; ++v) {
    int m = v + 8 * half;
    stateH[((size_t)dir * NB + b0 + m) * HID + hcol] = hreg[v];
    stateC[((size_t)dir * NB + b0 + m) * HID + hcol] = creg[v];
  }
  if (!L0 && last_chunk) {
#pragma unroll
    for (int v = 0; v < 8; ++v) {
      int m = v + 8 * half;
      out[(size_t)(b0 + m) * (2 * HID) + dir * HID + hcol] = hreg[v];
    }
  }
}

// ---------------- host ----------------
extern "C" void kernel_launch(void* const* d_in, const int* in_sizes, int n_in,
                              void* d_out, int out_size, void* d_ws,
                              size_t ws_size, hipStream_t stream) {
  const float* x        = (const float*)d_in[0];
  const float* w_ih_l0  = (const float*)d_in[1];
  const float* w_hh_l0  = (const float*)d_in[2];
  const float* b_l0     = (const float*)d_in[3];
  const float* w_ih_l0r = (const float*)d_in[4];
  const float* w_hh_l0r = (const float*)d_in[5];
  const float* b_l0r    = (const float*)d_in[6];
  const float* w_ih_l1  = (const float*)d_in[7];
  const float* w_hh_l1  = (const float*)d_in[8];
  const float* b_l1     = (const float*)d_in[9];
  const float* w_ih_l1r = (const float*)d_in[10];
  const float* w_hh_l1r = (const float*)d_in[11];
  const float* b_l1r    = (const float*)d_in[12];
  float* out = (float*)d_out;

  // workspace layout (bytes)
  char* w = (char*)d_ws;
  unsigned short* wih0  = (unsigned short*)(w + 0);          // 512*64  bf16
  unsigned short* wih0r = (unsigned short*)(w + 65536);
  unsigned short* wih1  = (unsigned short*)(w + 131072);     // 512*256 bf16
  unsigned short* wih1r = (unsigned short*)(w + 393216);
  unsigned short* whh0  = (unsigned short*)(w + 655360);     // 512*128 bf16
  unsigned short* whh0r = (unsigned short*)(w + 786432);
  unsigned short* whh1  = (unsigned short*)(w + 917504);
  unsigned short* whh1r = (unsigned short*)(w + 1048576);
  float* stH = (float*)(w + 1179648);                         // [2][256][128] f32
  float* stC = (float*)(w + 1441792);
  unsigned short* pre = (unsigned short*)(w + 1703936);       // [2][TC][256][512] bf16
  unsigned short* x1  = (unsigned short*)(w + 1703936 + (size_t)2 * TC * NB * G4 * 2);

  auto cvt = [&](const float* s, unsigned short* d, int n) {
    cvt_f32_bf16<<<dim3((n + 255) / 256), dim3(256), 0, stream>>>(s, d, n);
  };
  cvt(w_ih_l0, wih0, G4 * 64);
  cvt(w_ih_l0r, wih0r, G4 * 64);
  cvt(w_ih_l1, wih1, G4 * 256);
  cvt(w_ih_l1r, wih1r, G4 * 256);
  cvt(w_hh_l0, whh0, G4 * HID);
  cvt(w_hh_l0r, whh0r, G4 * HID);
  cvt(w_hh_l1, whh1, G4 * HID);
  cvt(w_hh_l1r, whh1r, G4 * HID);

  const int NC = TSEQ / TC;
  const dim3 pgrid(TC * 16, 2), pblk(256);
  const dim3 rgrid(16, 2), rblk(256);

  // layer 0 (input x, writes x1 as bf16)
  for (int ch = 0; ch < NC; ++ch) {
    proj_kernel<64, false><<<pgrid, pblk, 0, stream>>>(
        (const void*)x, (const unsigned*)wih0, (const unsigned*)wih0r, b_l0,
        b_l0r, pre, ch * TC);
    recur_kernel<true><<<rgrid, rblk, 0, stream>>>(
        pre, (const unsigned*)whh0, (const unsigned*)whh0r, stH, stC, x1,
        nullptr, ch * TC, 0);
  }
  // layer 1 (input x1 bf16, final h -> out)
  for (int ch = 0; ch < NC; ++ch) {
    proj_kernel<256, true><<<pgrid, pblk, 0, stream>>>(
        (const void*)x1, (const unsigned*)wih1, (const unsigned*)wih1r, b_l1,
        b_l1r, pre, ch * TC);
    recur_kernel<false><<<rgrid, rblk, 0, stream>>>(
        pre, (const unsigned*)whh1, (const unsigned*)whh1r, stH, stC, nullptr,
        out, ch * TC, ch == NC - 1);
  }
}
